// DotProductAttention_19146964206015
// MI455X (gfx1250) — matrix operand order repaired and checked by reading, not compiled, not run on hardware
//
#include <hip/hip_runtime.h>

// MI455X / gfx1250, wave32. WMMA bf16 16x16x32 + async global->LDS staging.

#define D_MODEL 1024
#define D_HEAD  128
#define SEQ     4096
#define BATCH   4

typedef __bf16 bf16;
typedef __attribute__((ext_vector_type(16))) __bf16 v16bf;
typedef __attribute__((ext_vector_type(8)))  __bf16 v8bf;
typedef __attribute__((ext_vector_type(8)))  float  v8f;
typedef __attribute__((ext_vector_type(4)))  float  v4f;

// ---------------------------------------------------------------------------
// Phase 0: convert the three [d_model, d_head] f32 weight matrices into
// bf16 transposed [d_head, d_model] so projection B-fragments are contiguous.
// ---------------------------------------------------------------------------
__global__ __launch_bounds__(256) void prep_w_kernel(
    const float* __restrict__ wq, const float* __restrict__ wk,
    const float* __restrict__ wv, bf16* __restrict__ wt /* 3 x [128][1024] */) {
  int idx = blockIdx.x * 256 + threadIdx.x;            // 0 .. 3*131072-1
  int which = idx / (D_MODEL * D_HEAD);
  int r     = idx % (D_MODEL * D_HEAD);
  int d = r / D_MODEL;
  int m = r % D_MODEL;
  const float* w = (which == 0) ? wq : (which == 1) ? wk : wv;
  wt[(size_t)which * D_HEAD * D_MODEL + (size_t)d * D_MODEL + m] =
      (bf16)w[(size_t)m * D_HEAD + d];
}

// ---------------------------------------------------------------------------
// Phase 1: projection  out[b,s,d] = sum_m X[b,s,m] * W[m,d], bf16 output.
// Block = 128 threads (4 waves): one 16-row s-tile x eight 16-col d-tiles.
// transpose_out=1 writes V as [B,128,S] for contiguous PV B-fragment loads.
// ---------------------------------------------------------------------------
__global__ __launch_bounds__(128) void proj_kernel(
    const float* __restrict__ x,   // [B, S, D_MODEL] f32
    const bf16*  __restrict__ wt,  // [D_HEAD, D_MODEL] bf16 (transposed)
    bf16* __restrict__ out, int transpose_out) {
  const int lane = threadIdx.x & 31;
  const int wave = threadIdx.x >> 5;    // 0..3
  const int half = lane >> 4;           // 0/1
  const int l16  = lane & 15;
  const int s0   = blockIdx.x * 16;
  const int b    = blockIdx.y;
  const int nt0  = wave * 2;            // this wave's first d-tile

  const float* xrow = x + (size_t)(b * SEQ + s0 + l16) * D_MODEL;

  v8f acc0 = {}; v8f acc1 = {};

  for (int kc = 0; kc < D_MODEL; kc += 32) {
    // A fragment (16x32 bf16): lane<16 -> els0-15:K0-15
    //                          lane>=16 -> els0-15:K16-31
    v16bf a;
    {
      const float* p0 = xrow + kc + half * 16;
      const float* p1 = xrow + kc + half * 16 + 8;
      v4f f0 = *(const v4f*)(p0);
      v4f f1 = *(const v4f*)(p0 + 4);
      v4f f2 = *(const v4f*)(p1);
      v4f f3 = *(const v4f*)(p1 + 4);
#pragma unroll
      for (int i = 0; i < 4; ++i) {
        a[i]      = (bf16)f0[i];
        a[4 + i]  = (bf16)f1[i];
        a[8 + i]  = (bf16)f2[i];
        a[12 + i] = (bf16)f3[i];
      }
    }
    // B fragment (32x16): lane el i -> K = i + half*16, N = l16
    v16bf b0 = *(const v16bf*)(wt + (size_t)(nt0 * 16 + l16) * D_MODEL + kc + half * 16);
    v16bf b1 = *(const v16bf*)(wt + (size_t)((nt0 + 1) * 16 + l16) * D_MODEL + kc + half * 16);
    acc0 = __builtin_amdgcn_wmma_f32_16x16x32_bf16(false, a, false, b0, (short)0, acc0, false, false);
    acc1 = __builtin_amdgcn_wmma_f32_16x16x32_bf16(false, a, false, b1, (short)0, acc1, false, false);
  }

  // C/D layout: vgpr j -> row m = j + half*8, col n = l16
#pragma unroll
  for (int j = 0; j < 8; ++j) {
    int s = s0 + j + half * 8;
    int d0 = nt0 * 16 + l16;
    int d1 = (nt0 + 1) * 16 + l16;
    bf16 v0 = (bf16)acc0[j];
    bf16 v1 = (bf16)acc1[j];
    if (transpose_out) {
      out[((size_t)b * D_HEAD + d0) * SEQ + s] = v0;
      out[((size_t)b * D_HEAD + d1) * SEQ + s] = v1;
    } else {
      out[(size_t)(b * SEQ + s) * D_HEAD + d0] = v0;
      out[(size_t)(b * SEQ + s) * D_HEAD + d1] = v1;
    }
  }
}

// ---------------------------------------------------------------------------
// Async global->LDS staging of one 32-key K/V tile (16 KB) by 256 threads.
// Each thread issues 4 x global_load_async_to_lds_b128 (tracked by ASYNCcnt).
// K tile: rows kb..kb+31 of [S,128] bf16 -> contiguous 8 KB block.
// V tile: 128 rows (d) x 32 keys from [128,S] bf16 -> 128 x 64 B strided.
// ---------------------------------------------------------------------------
__device__ __forceinline__ void stage_tile(const bf16* kbase, const bf16* vbase,
                                           bf16* skbuf, bf16* svbuf,
                                           int kb, int tid) {
#pragma unroll
  for (int u = 0; u < 2; ++u) {
    int chunk = tid * 2 + u;                         // 0..511 (16B chunks)
    // K: whole tile contiguous at kbase + kb*128
    {
      unsigned dst = (unsigned)(uintptr_t)(skbuf + chunk * 8);
      unsigned long long src =
          (unsigned long long)(uintptr_t)(kbase + (size_t)kb * D_HEAD + chunk * 8);
      asm volatile("global_load_async_to_lds_b128 %0, %1, off"
                   :: "v"(dst), "v"(src) : "memory");
    }
    // V: row = d, 4 x 16B per row
    {
      int row = chunk >> 2, part = chunk & 3;
      unsigned dst = (unsigned)(uintptr_t)(svbuf + row * 32 + (((part & 1) << 1) | (part >> 1)) * 8);
      unsigned long long src =
          (unsigned long long)(uintptr_t)(vbase + (size_t)row * SEQ + kb + part * 8);
      asm volatile("global_load_async_to_lds_b128 %0, %1, off"
                   :: "v"(dst), "v"(src) : "memory");
    }
  }
}

// ---------------------------------------------------------------------------
// Phase 2: flash attention. 8 waves/block, 1 wave per 16-query tile; the
// block cooperatively double-buffers the shared 32-key K/V tile in LDS via
// async copies. Scores computed TRANSPOSED (S^T = K*Q^T) so exp() outputs
// drop straight into the P A-fragment layout for P*V (zero lane shuffles).
// ---------------------------------------------------------------------------
__global__ __launch_bounds__(256) void attn_kernel(
    const bf16* __restrict__ qbf,  // [B, S, 128] bf16
    const bf16* __restrict__ kbf,  // [B, S, 128] bf16
    const bf16* __restrict__ vt,   // [B, 128, S] bf16 (transposed)
    float* __restrict__ out) {     // [B, S, 128] f32
  __shared__ alignas(64) bf16 sk[2][32 * D_HEAD];    // 2 x 8 KB
  __shared__ alignas(64) bf16 sv[2][D_HEAD * 32];    // 2 x 8 KB

  const int tid  = threadIdx.x;
  const int lane = tid & 31;
  const int wave = tid >> 5;           // 0..7
  const int half = lane >> 4;
  const int l16  = lane & 15;
  const int b    = blockIdx.y;
  const int s0   = (blockIdx.x * 8 + wave) * 16;

  // Q^T B-fragments: el i -> K(d)=c*32+half*16+i, N(q)=l16
  v16bf qf[4];
  const bf16* qrow = qbf + (size_t)(b * SEQ + s0 + l16) * D_HEAD;
#pragma unroll
  for (int c = 0; c < 4; ++c)
    qf[c] = *(const v16bf*)(qrow + c * 32 + half * 16);

  const float scale = 0.08838834764831845f * 1.4426950408889634f; // rsqrt(128)*log2(e)

  float m_i = -__builtin_inff();
  float l_i = 0.0f;
  v8f zero = {};
  v8f o[8];
#pragma unroll
  for (int n = 0; n < 8; ++n) o[n] = zero;

  const bf16* kbase = kbf + (size_t)b * SEQ * D_HEAD;
  const bf16* vbase = vt  + (size_t)b * D_HEAD * SEQ;

  // prologue: stage first tile
  stage_tile(kbase, vbase, sk[0], sv[0], 0, tid);
  asm volatile("s_wait_asynccnt 0x0" ::: "memory");
  __syncthreads();

  for (int kb = 0; kb < SEQ; kb += 32) {
    const int buf = (kb >> 5) & 1;
    // prefetch next tile into the other buffer (overlaps with compute below)
    if (kb + 32 < SEQ)
      stage_tile(kbase, vbase, sk[buf ^ 1], sv[buf ^ 1], kb + 32, tid);

    const bf16* skb = sk[buf];
    const bf16* svb = sv[buf];

    // --- scores: S^T[key, q] for two 16-key subtiles, K from LDS ---
    v8f st[2];
#pragma unroll
    for (int t = 0; t < 2; ++t) {
      st[t] = zero;
      const bf16* krow = skb + (t * 16 + l16) * D_HEAD;
#pragma unroll
      for (int c = 0; c < 4; ++c) {
        v8bf lo = *(const v8bf*)(krow + c * 32 + half * 16);
        v8bf hi = *(const v8bf*)(krow + c * 32 + half * 16 + 8);
        v16bf ka;
#pragma unroll
        for (int i = 0; i < 8; ++i) { ka[i] = lo[i]; ka[8 + i] = hi[i]; }
        st[t] = __builtin_amdgcn_wmma_f32_16x16x32_bf16(false, ka, false, qf[c],
                                                        (short)0, st[t], false, false);
      }
    }

    // --- online softmax; this lane holds query q=l16, keys {t*16 + half*8 + j} ---
    float cmax = m_i;
#pragma unroll
    for (int t = 0; t < 2; ++t)
#pragma unroll
      for (int j = 0; j < 8; ++j) {
        float sc = st[t][j] * scale;
        st[t][j] = sc;
        cmax = fmaxf(cmax, sc);
      }
    cmax = fmaxf(cmax, __shfl_xor(cmax, 16, 32));
    float m_new = cmax;
    float alpha = exp2f(m_i - m_new);

    float rowsum = 0.0f;
    v16bf pa;                       // P A-fragment, packed directly
#pragma unroll
    for (int t = 0; t < 2; ++t)
#pragma unroll
      for (int j = 0; j < 8; ++j) {
        float p = exp2f(st[t][j] - m_new);
        rowsum += p;
        pa[t * 8 + j] = (bf16)p;
      }
    rowsum += __shfl_xor(rowsum, 16, 32);
    l_i = alpha * l_i + rowsum;
    m_i = m_new;

    // rescale O: accumulator row (vgpr j) is query j + half*8 -> broadcast alpha
#pragma unroll
    for (int j = 0; j < 8; ++j) {
      float aj = __shfl(alpha, j + half * 8, 32);
#pragma unroll
      for (int n = 0; n < 8; ++n) o[n][j] *= aj;
    }

    // --- P * V, V from LDS: B el i -> K(key)=half*8+i (i<8) / 16+half*8+i-8 (i>=8), N(d)=n*16+l16 ---
#pragma unroll
    for (int n = 0; n < 8; ++n) {
      v16bf vb = *(const v16bf*)(svb + (n * 16 + l16) * 32 + half * 16);
      o[n] = __builtin_amdgcn_wmma_f32_16x16x32_bf16(false, pa, false, vb,
                                                     (short)0, o[n], false, false);
    }

    // copies for next iteration must land; LDS reads of this tile are done
    asm volatile("s_wait_asynccnt 0x0" ::: "memory");
    __syncthreads();
  }

  // normalize and store: O vgpr j -> q = s0 + j + half*8, d = n*16 + l16
  float inv = 1.0f / l_i;
#pragma unroll
  for (int j = 0; j < 8; ++j) {
    float invj = __shfl(inv, j + half * 8, 32);
    float* orow = out + (size_t)(b * SEQ + s0 + j + half * 8) * D_HEAD + l16;
#pragma unroll
    for (int n = 0; n < 8; ++n)
      orow[n * 16] = o[n][j] * invj;
  }
}

// ---------------------------------------------------------------------------
extern "C" void kernel_launch(void* const* d_in, const int* in_sizes, int n_in,
                              void* d_out, int out_size, void* d_ws, size_t ws_size,
                              hipStream_t stream) {
  const float* xq = (const float*)d_in[0];
  const float* xk = (const float*)d_in[1];
  const float* xv = (const float*)d_in[2];
  const float* wq = (const float*)d_in[3];
  const float* wk = (const float*)d_in[4];
  const float* wv = (const float*)d_in[5];

  // workspace layout (bf16): 3 transposed weights + Q + K + V^T  (~13.4 MB)
  char* ws = (char*)d_ws;
  const size_t wt_bytes  = 3ull * D_HEAD * D_MODEL * sizeof(bf16);
  const size_t qkv_bytes = (size_t)BATCH * SEQ * D_HEAD * sizeof(bf16);
  bf16* wt  = (bf16*)ws;
  bf16* qbf = (bf16*)(ws + wt_bytes);
  bf16* kbf = (bf16*)(ws + wt_bytes + qkv_bytes);
  bf16* vtp = (bf16*)(ws + wt_bytes + 2 * qkv_bytes);

  prep_w_kernel<<<dim3((3 * D_MODEL * D_HEAD) / 256), 256, 0, stream>>>(wq, wk, wv, wt);

  dim3 pgrid(SEQ / 16, BATCH);
  proj_kernel<<<pgrid, 128, 0, stream>>>(xq, wt,                            qbf, 0);
  proj_kernel<<<pgrid, 128, 0, stream>>>(xk, wt + (size_t)D_HEAD * D_MODEL, kbf, 0);
  proj_kernel<<<pgrid, 128, 0, stream>>>(xv, wt + 2ull * D_HEAD * D_MODEL,  vtp, 1);

  dim3 agrid(SEQ / 128, BATCH);   // 8 waves/block, 1 wave per 16-query tile
  attn_kernel<<<agrid, 256, 0, stream>>>(qbf, kbf, vtp, (float*)d_out);
}
